// MultiQueryAttention_5832565588415
// MI455X (gfx1250) — matrix operand order repaired and checked by reading, not compiled, not run on hardware
//
#include <hip/hip_runtime.h>
#include <hip/hip_bf16.h>

// ---------------- common types ----------------
typedef __attribute__((ext_vector_type(16))) __bf16 v16bf;
typedef __attribute__((ext_vector_type(8)))  float  v8f;
typedef unsigned int u32x4 __attribute__((ext_vector_type(4)));
typedef int          i32x8 __attribute__((ext_vector_type(8)));
typedef int          i32x4 __attribute__((ext_vector_type(4)));

union AFrag { uint4 u2[2]; v16bf v; };           // 16x32 bf16 A operand (8 VGPRs/lane)
union BFrag { unsigned int w[8]; v16bf v; };     // 32x16 bf16 B operand (8 VGPRs/lane)
union PackP { __bf16 h[2]; unsigned int u; };

static constexpr int Bn  = 2;
static constexpr int Tn  = 2048;
static constexpr int Cn  = 1024;   // DIM
static constexpr int Hn  = 16;
static constexpr int Dn  = 64;     // HEAD_DIM
static constexpr int BT  = Bn * Tn; // 4096

// ---------------- elementwise helpers ----------------
__global__ void convert_x_kernel(const float* __restrict__ x, __bf16* __restrict__ xb, int n) {
    int i = blockIdx.x * blockDim.x + threadIdx.x;
    if (i < n) xb[i] = (__bf16)x[i];
}

// W (K x N, row-major f32) -> pair-packed B layout: out[k2*N + n] packs rows (2k2, 2k2+1)
__global__ void pack_w_kernel(const float* __restrict__ W, unsigned int* __restrict__ out,
                              int K, int N) {
    int i = blockIdx.x * blockDim.x + threadIdx.x;
    int total = (K >> 1) * N;
    if (i >= total) return;
    int k2 = i / N, n = i - k2 * N;
    PackP p;
    p.h[0] = (__bf16)W[(size_t)(2 * k2) * N + n];
    p.h[1] = (__bf16)W[(size_t)(2 * k2 + 1) * N + n];
    out[i] = p.u;
}

// ---------------- GEMM: C = A(MxK bf16) * Bpacked(KxN) + bias, TDM-staged B ----------------
// Block = 8 waves = 8 consecutive M-tiles sharing one 64-wide N strip.
// B strip staged into LDS in two K-halves via the Tensor Data Mover.
__global__ __launch_bounds__(256)
void gemm_bf16_kernel(const __bf16* __restrict__ A, const unsigned int* __restrict__ Bp,
                      const float* __restrict__ bias, float* __restrict__ C,
                      int M, int N, int K) {
    __shared__ unsigned int bsh[256 * 64];   // 64 KB: 256 pair-rows x 64 cols
    int wv    = threadIdx.x >> 5;
    int lane  = threadIdx.x & 31;
    int nblk  = N >> 6;
    int bm    = blockIdx.x / nblk;
    int bn    = blockIdx.x - bm * nblk;
    int m0    = bm * 128 + wv * 16;
    int n0    = bn * 64;
    int lhalf = lane >> 4;
    int l16   = lane & 15;
    int abase = lhalf ? 8 : 0;

    v8f acc[4] = {};
    const __bf16* arow = A + (size_t)(m0 + l16) * K;
    int rowsPerHalf = K >> 2;                // pair-rows staged per half (256 for K=1024)

    for (int half = 0; half < 2; ++half) {
        int kh = half * (K >> 1);
        // ---- stage B pair-rows [kh/2, kh/2 + rowsPerHalf) x cols [n0, n0+64) into LDS ----
#if __has_builtin(__builtin_amdgcn_tensor_load_to_lds) && __has_builtin(__builtin_amdgcn_s_wait_tensorcnt)
        if (wv == 0) {
            unsigned long long ga =
                (unsigned long long)(const void*)(Bp + (size_t)(kh >> 1) * N + n0);
            unsigned int ldsOff = (unsigned int)(unsigned long long)(const void*)&bsh[0];
            u32x4 g0;
            g0[0] = 1u;                                  // count=1, user descriptor
            g0[1] = ldsOff;                              // lds_addr
            g0[2] = (unsigned int)(ga & 0xFFFFFFFFull);  // global_addr[31:0]
            g0[3] = (unsigned int)((ga >> 32) & 0x01FFFFFFull) | (2u << 30); // addr hi | type=2
            unsigned int td0 = (unsigned int)N;          // tensor_dim0 (elements)
            unsigned int td1 = (unsigned int)rowsPerHalf;// tensor_dim1 (rows)
            unsigned long long st0 = (unsigned long long)N; // tensor_dim0_stride
            i32x8 g1;
            g1[0] = (int)(2u << 16);                               // data_size=4B
            g1[1] = (int)((td0 & 0xFFFFu) << 16);                  // tensor_dim0 lo16
            g1[2] = (int)((td0 >> 16) | ((td1 & 0xFFFFu) << 16));  // dim0 hi16 | dim1 lo16
            g1[3] = (int)((td1 >> 16) | (64u << 16));              // dim1 hi16 | tile_dim0=64
            g1[4] = (int)td1;                                      // tile_dim1 | tile_dim2=0
            g1[5] = (int)(unsigned int)(st0 & 0xFFFFFFFFull);      // stride0 lo32
            g1[6] = (int)(unsigned int)(st0 >> 32);                // stride0 hi16 | stride1 lo16
            g1[7] = 0;                                             // stride1 hi
            i32x4 gz4 = {};
            i32x8 gz8 = {};
            __builtin_amdgcn_tensor_load_to_lds(g0, g1, gz4, gz4, gz8, 0);
            __builtin_amdgcn_s_wait_tensorcnt(0);
        }
        __syncthreads();
#else
        // Fallback: cooperative copy (all 256 threads)
        {
            const unsigned int* src = Bp + (size_t)(kh >> 1) * N + n0;
            for (int i = threadIdx.x; i < rowsPerHalf * 64; i += 256) {
                int r = i >> 6, c = i & 63;
                bsh[i] = src[(size_t)r * N + c];
            }
        }
        __syncthreads();
#endif
        // ---- compute over this K half ----
        for (int k0 = kh; k0 < kh + (K >> 1); k0 += 32) {
            __builtin_prefetch(arow + k0 + 64, 0, 1);
            AFrag a;
            a.u2[0] = *(const uint4*)(arow + k0 + abase);
            a.u2[1] = *(const uint4*)(arow + k0 + 16 + abase);
            int krow = ((k0 - kh) >> 1) + lhalf * 4;
#pragma unroll
            for (int nt = 0; nt < 4; ++nt) {
                BFrag b;
#pragma unroll
                for (int v = 0; v < 8; ++v) b.w[v] = bsh[(krow + v + (v & 4)) * 64 + nt * 16 + l16];
                acc[nt] = __builtin_amdgcn_wmma_f32_16x16x32_bf16(
                    false, a.v, false, b.v, (short)0, acc[nt], false, false);
            }
        }
        __syncthreads();   // protect LDS before next half overwrites
    }

#pragma unroll
    for (int nt = 0; nt < 4; ++nt) {
        int col = n0 + nt * 16 + l16;
        float bvv = bias[col];
#pragma unroll
        for (int r = 0; r < 8; ++r) {
            int orow = m0 + r + 8 * lhalf;
            C[(size_t)orow * N + col] = acc[nt][r] + bvv;
        }
    }
}

// ---------------- RoPE on Q (+ fold 1/sqrt(64)), output bf16 (b,t,h*64+d) ----------------
__global__ void rope_q_kernel(const float* __restrict__ Qraw, __bf16* __restrict__ Qb, int n) {
    int i = blockIdx.x * blockDim.x + threadIdx.x;   // over BT*H*16
    if (i >= n) return;
    int j  = i & 15;
    int h  = (i >> 4) & 15;
    int bt = i >> 8;
    int t  = bt & (Tn - 1);
    const float* q = Qraw + (size_t)bt * Cn + h * Dn;
    __bf16*      o = Qb   + (size_t)bt * Cn + h * Dn;
    float x0 = q[j], x1 = q[j + 16], x2 = q[j + 32], x3 = q[j + 48];
    float th0 = __powf(10000.f, -(float)j * (1.f / 32.f));
    float th1 = __powf(10000.f, -(float)(j + 16) * (1.f / 32.f));
    float a0 = (float)t * th0, a1 = (float)t * th1;
    const float sc = 0.125f;   // 1/sqrt(HEAD_DIM)
    o[j]      = (__bf16)((x0 * __cosf(a0) - x1 * __sinf(a0)) * sc);
    o[j + 16] = (__bf16)((x1 * __cosf(a1) + x0 * __sinf(a1)) * sc);
    o[j + 32] = (__bf16)(x2 * sc);
    o[j + 48] = (__bf16)(x3 * sc);
}

// ---------------- RoPE on K + transpose + pair-pack: KB[b][d2][t] ----------------
__global__ void rope_pack_k_kernel(const float* __restrict__ Kraw, unsigned int* __restrict__ KB, int n) {
    int i = blockIdx.x * blockDim.x + threadIdx.x;   // over B*T
    if (i >= n) return;
    int b = i / Tn, t = i - b * Tn;
    const float* k = Kraw + (size_t)i * Dn;
    float kv[64];
#pragma unroll
    for (int j = 0; j < 16; ++j) {
        float th0 = __powf(10000.f, -(float)j * (1.f / 32.f));
        float th1 = __powf(10000.f, -(float)(j + 16) * (1.f / 32.f));
        float a0 = (float)t * th0, a1 = (float)t * th1;
        kv[j]      = k[j] * __cosf(a0) - k[j + 16] * __sinf(a0);
        kv[j + 16] = k[j + 16] * __cosf(a1) + k[j] * __sinf(a1);
    }
#pragma unroll
    for (int d = 32; d < 64; ++d) kv[d] = k[d];
    unsigned int* out = KB + (size_t)b * 32 * Tn + t;
#pragma unroll
    for (int d2 = 0; d2 < 32; ++d2) {
        PackP p;
        p.h[0] = (__bf16)kv[2 * d2];
        p.h[1] = (__bf16)kv[2 * d2 + 1];
        out[(size_t)d2 * Tn] = p.u;
    }
}

// ---------------- pack V along time pairs: VB[b][t2][d] ----------------
__global__ void pack_v_kernel(const float* __restrict__ Vraw, unsigned int* __restrict__ VB, int n) {
    int i = blockIdx.x * blockDim.x + threadIdx.x;   // over B*(T/2)*64
    if (i >= n) return;
    int d   = i & 63;
    int bt2 = i >> 6;
    int b   = bt2 / (Tn / 2);
    int t2  = bt2 - b * (Tn / 2);
    size_t base = ((size_t)b * Tn + 2 * t2) * Dn + d;
    PackP p;
    p.h[0] = (__bf16)Vraw[base];
    p.h[1] = (__bf16)Vraw[base + Dn];
    VB[i] = p.u;
}

// ---------------- flash attention: one wave per 16-row Q strip per (b,h) ----------------
__global__ __launch_bounds__(128)
void flash_attn_kernel(const __bf16* __restrict__ Qb, const unsigned int* __restrict__ KB,
                       const unsigned int* __restrict__ VB, __bf16* __restrict__ Ao) {
    __shared__ __bf16 lds[4][16 * 32];
    int wv   = threadIdx.x >> 5;
    int lane = threadIdx.x & 31;
    int gid  = blockIdx.x * 4 + wv;
    if (gid >= Bn * Hn * (Tn / 16)) return;
    int bh = gid >> 7;                  // T/16 = 128 strips per (b,h)
    int q0 = (gid & 127) << 4;
    int b = bh >> 4, h = bh & 15;
    int lhalf = lane >> 4;
    int l16   = lane & 15;
    int abase = lhalf ? 8 : 0;

    // Q A-fragments: rows q0..q0+15, d split into two 16x32 K-chunks
    const __bf16* qp = Qb + (size_t)(b * Tn + q0 + l16) * Cn + h * Dn;
    AFrag aq[2];
    aq[0].u2[0] = *(const uint4*)(qp + 0  + abase);
    aq[0].u2[1] = *(const uint4*)(qp + 16 + abase);
    aq[1].u2[0] = *(const uint4*)(qp + 32 + abase);
    aq[1].u2[1] = *(const uint4*)(qp + 48 + abase);

    v8f acc[4] = {};
    float m[8], s[8];
#pragma unroll
    for (int r = 0; r < 8; ++r) { m[r] = -__builtin_inff(); s[r] = 0.f; }

    const unsigned int* kbase = KB + (size_t)b * 32 * Tn;
    const unsigned int* vbase = VB + (size_t)b * (Tn / 2) * Dn;
    __bf16* pl = &lds[wv][0];

    int nkb = (q0 + 15) / 32 + 1;
    for (int kb = 0; kb < nkb; ++kb) {
        int t0 = kb * 32;
        // ---- scores: S(16 x 32) = Q(16x64) * K^T(64x32), two 16x16 C tiles ----
        v8f c0 = {}, c1 = {};
#pragma unroll
        for (int kc = 0; kc < 2; ++kc) {
            const unsigned int* kp = kbase + (size_t)(kc * 16 + lhalf * 4) * Tn + t0;
            BFrag bk0, bk1;
#pragma unroll
            for (int v = 0; v < 8; ++v) {
                bk0.w[v] = kp[(size_t)(v + (v & 4)) * Tn + l16];
                bk1.w[v] = kp[(size_t)(v + (v & 4)) * Tn + 16 + l16];
            }
            c0 = __builtin_amdgcn_wmma_f32_16x16x32_bf16(false, aq[kc].v, false, bk0.v, (short)0, c0, false, false);
            c1 = __builtin_amdgcn_wmma_f32_16x16x32_bf16(false, aq[kc].v, false, bk1.v, (short)0, c1, false, false);
        }
        // ---- causal mask ----
        if (t0 + 31 > q0) {
#pragma unroll
            for (int r = 0; r < 8; ++r) {
                int rowt = q0 + r + 8 * lhalf;
                if (t0 + l16 > rowt)      c0[r] = -1e30f;
                if (t0 + 16 + l16 > rowt) c1[r] = -1e30f;
            }
        }
        // ---- online softmax ----
#pragma unroll
        for (int r = 0; r < 8; ++r) {
            float mx = fmaxf(c0[r], c1[r]);
#pragma unroll
            for (int off = 8; off >= 1; off >>= 1) mx = fmaxf(mx, __shfl_xor(mx, off, 32));
            float mnew = fmaxf(m[r], mx);
            float sc   = __expf(m[r] - mnew);
            float p0 = __expf(c0[r] - mnew);
            float p1 = __expf(c1[r] - mnew);
            c0[r] = p0; c1[r] = p1;
            float rs = p0 + p1;
#pragma unroll
            for (int off = 8; off >= 1; off >>= 1) rs += __shfl_xor(rs, off, 32);
            s[r] = s[r] * sc + rs;
            m[r] = mnew;
            acc[0][r] *= sc; acc[1][r] *= sc; acc[2][r] *= sc; acc[3][r] *= sc;
        }
        // ---- P: C-layout f32 -> A-layout bf16 via wave-private LDS ----
#pragma unroll
        for (int r = 0; r < 8; ++r) {
            int row = r + 8 * lhalf;
            pl[row * 32 + l16]      = (__bf16)c0[r];
            pl[row * 32 + 16 + l16] = (__bf16)c1[r];
        }
        asm volatile("s_wait_dscnt 0" ::: "memory");
        AFrag ap;
        ap.u2[0] = *(const uint4*)(pl + l16 * 32 + abase);
        ap.u2[1] = *(const uint4*)(pl + l16 * 32 + 16 + abase);
        // ---- O += P(16x32) * V(32x64) ----
        const unsigned int* vp = vbase + (size_t)(t0 / 2 + lhalf * 4) * Dn;
#pragma unroll
        for (int nt = 0; nt < 4; ++nt) {
            BFrag bv;
#pragma unroll
            for (int v = 0; v < 8; ++v) bv.w[v] = vp[(size_t)(v + (v & 4)) * Dn + nt * 16 + l16];
            acc[nt] = __builtin_amdgcn_wmma_f32_16x16x32_bf16(
                false, ap.v, false, bv.v, (short)0, acc[nt], false, false);
        }
    }

    // ---- finalize: divide by row sum, store bf16 (b,t,h*64+d) ----
#pragma unroll
    for (int r = 0; r < 8; ++r) {
        float inv = 1.0f / s[r];
        int t = q0 + r + 8 * lhalf;
        __bf16* op = Ao + (size_t)(b * Tn + t) * Cn + h * Dn + l16;
        op[0]  = (__bf16)(acc[0][r] * inv);
        op[16] = (__bf16)(acc[1][r] * inv);
        op[32] = (__bf16)(acc[2][r] * inv);
        op[48] = (__bf16)(acc[3][r] * inv);
    }
}

// ---------------- launch ----------------
extern "C" void kernel_launch(void* const* d_in, const int* in_sizes, int n_in,
                              void* d_out, int out_size, void* d_ws, size_t ws_size,
                              hipStream_t stream) {
    const float* x  = (const float*)d_in[0];
    const float* Wq = (const float*)d_in[1];
    const float* bq = (const float*)d_in[2];
    const float* Wk = (const float*)d_in[3];
    const float* bk = (const float*)d_in[4];
    const float* Wv = (const float*)d_in[5];
    const float* bv = (const float*)d_in[6];
    const float* Wo = (const float*)d_in[7];
    const float* bo = (const float*)d_in[8];
    float* out = (float*)d_out;

    char* ws = (char*)d_ws;
    size_t off = 0;
    auto take = [&](size_t bytes) -> void* {
        void* p = (void*)(ws + off);
        off += (bytes + 255) & ~(size_t)255;
        return p;
    };

    __bf16*       Xb   = (__bf16*)      take((size_t)BT * Cn * 2);
    unsigned int* WqB  = (unsigned int*)take((size_t)(Cn / 2) * Cn * 4);
    unsigned int* WkB  = (unsigned int*)take((size_t)(Cn / 2) * Dn * 4);
    unsigned int* WvB  = (unsigned int*)take((size_t)(Cn / 2) * Dn * 4);
    unsigned int* WoB  = (unsigned int*)take((size_t)(Cn / 2) * Cn * 4);
    float*        Qraw = (float*)       take((size_t)BT * Cn * 4);
    float*        Kraw = (float*)       take((size_t)BT * Dn * 4);
    float*        Vraw = (float*)       take((size_t)BT * Dn * 4);
    __bf16*       Qbuf = (__bf16*)      take((size_t)BT * Cn * 2);
    unsigned int* KBp  = (unsigned int*)take((size_t)Bn * 32 * Tn * 4);
    unsigned int* VBp  = (unsigned int*)take((size_t)Bn * (Tn / 2) * Dn * 4);
    __bf16*       Ao   = (__bf16*)      take((size_t)BT * Cn * 2);

    // 1) x -> bf16
    {
        int n = BT * Cn;
        convert_x_kernel<<<(n + 255) / 256, 256, 0, stream>>>(x, Xb, n);
    }
    // 2) pack all weights into WMMA-B pair layout
    {
        int n = (Cn / 2) * Cn;
        pack_w_kernel<<<(n + 255) / 256, 256, 0, stream>>>(Wq, WqB, Cn, Cn);
        pack_w_kernel<<<(n + 255) / 256, 256, 0, stream>>>(Wo, WoB, Cn, Cn);
        int nk = (Cn / 2) * Dn;
        pack_w_kernel<<<(nk + 255) / 256, 256, 0, stream>>>(Wk, WkB, Cn, Dn);
        pack_w_kernel<<<(nk + 255) / 256, 256, 0, stream>>>(Wv, WvB, Cn, Dn);
    }
    // 3) projections (WMMA GEMMs, TDM-staged B). Blocks = (M/128) * (N/64).
    {
        int blocksQ = (BT / 128) * (Cn / 64);          // 512
        gemm_bf16_kernel<<<blocksQ, 256, 0, stream>>>(Xb, WqB, bq, Qraw, BT, Cn, Cn);
        int blocksK = (BT / 128) * (Dn / 64);          // 32
        gemm_bf16_kernel<<<blocksK, 256, 0, stream>>>(Xb, WkB, bk, Kraw, BT, Dn, Cn);
        gemm_bf16_kernel<<<blocksK, 256, 0, stream>>>(Xb, WvB, bv, Vraw, BT, Dn, Cn);
    }
    // 4) RoPE + layout packing
    {
        int nq = BT * Hn * 16;
        rope_q_kernel<<<(nq + 255) / 256, 256, 0, stream>>>(Qraw, Qbuf, nq);
        int nk = Bn * Tn;
        rope_pack_k_kernel<<<(nk + 255) / 256, 256, 0, stream>>>(Kraw, KBp, nk);
        int nv = Bn * (Tn / 2) * Dn;
        pack_v_kernel<<<(nv + 255) / 256, 256, 0, stream>>>(Vraw, VBp, nv);
    }
    // 5) flash attention (causal, online softmax)
    {
        int waves = Bn * Hn * (Tn / 16);               // 4096
        flash_attn_kernel<<<waves / 4, 128, 0, stream>>>(Qbuf, KBp, VBp, Ao);
    }
    // 6) output projection -> d_out (fp32)
    {
        int blocks = (BT / 128) * (Cn / 64);           // 512
        gemm_bf16_kernel<<<blocks, 256, 0, stream>>>(Ao, WoB, bo, out, BT, Cn, Cn);
    }
    (void)in_sizes; (void)n_in; (void)out_size; (void)ws_size;
}
